// net_crossInteraction_29961691857343
// MI455X (gfx1250) — matrix operand order repaired and checked by reading, not compiled, not run on hardware
//
#include <hip/hip_runtime.h>

// ---------------------------------------------------------------------------
// MI455X (gfx1250) cross-interaction network.
// All GEMMs: C = A(f16, MxK) x Bt(f16, NxK)^T via v_wmma_f32_16x16x32_f16.
// Operands are pre-converted to f16 (weights transpose-converted once; chained
// activations written as f16 by the GEMM epilogue). Each wave computes a
// 32x64 strip (2 row tiles x 4 col tiles): per K-step the hot loop is
// 12 b128 loads + 8 WMMA, with no f32->f16 conversion and no strided loads.
// ---------------------------------------------------------------------------

typedef __attribute__((ext_vector_type(16))) _Float16 v16h;
typedef __attribute__((ext_vector_type(8)))  _Float16 v8h;
typedef __attribute__((ext_vector_type(8)))  float    v8f;

constexpr int kB = 8;      // batch
constexpr int kN = 1000;   // protein nodes (40*25)
constexpr int kD = 256;    // feature dim
constexpr int kM = 56;     // drug nodes
constexpr int kMp = 64;    // drug nodes padded to a full 4-tile column strip

// ------------------------------ embedding ---------------------------------
__global__ void embed_kernel(const int* __restrict__ prot,
                             const float* __restrict__ table,
                             float* __restrict__ seq, float* __restrict__ x,
                             _Float16* __restrict__ seqh, _Float16* __restrict__ xh)
{
    int bi = blockIdx.x;            // 0..B*N-1
    int d  = threadIdx.x;           // 0..255
    float v = table[prot[bi] * kD + d];
    long  o = (long)bi * kD + d;
    seq[o] = v; x[o] = v;
    seqh[o] = (_Float16)v; xh[o] = (_Float16)v;
}

// ---------------------- conversion / transpose helpers ---------------------
__global__ void cvt_kernel(const float* __restrict__ in,
                           _Float16* __restrict__ out, long n)
{
    long g = (long)blockIdx.x * blockDim.x + threadIdx.x;
    if (g < n) out[g] = (_Float16)in[g];
}

__global__ void relu_h_kernel(const float* __restrict__ in,
                              _Float16* __restrict__ out, long n)
{
    long g = (long)blockIdx.x * blockDim.x + threadIdx.x;
    if (g < n) out[g] = (_Float16)fmaxf(in[g], 0.0f);
}

// dst[N][K] = (f16) src[K][N]   (weight transpose-convert, once per call)
__global__ void cvt_t_kernel(const float* __restrict__ src,
                             _Float16* __restrict__ dst, int K, int N)
{
    int n = blockIdx.x;
    for (int k = threadIdx.x; k < K; k += blockDim.x)
        dst[(long)n * K + k] = (_Float16)src[(long)k * N + n];
}

// dst[b][c][r] = src[b][r][c], cols == blockDim.x (f16 -> f16)
__global__ void transpose_h_kernel(const _Float16* __restrict__ src,
                                   _Float16* __restrict__ dst, int rows)
{
    int bi = blockIdx.x;                // b*rows + r
    int b = bi / rows, r = bi - b * rows;
    int c = threadIdx.x, cols = blockDim.x;
    dst[((long)b * cols + c) * rows + r] = src[((long)b * rows + r) * cols + c];
}

// ------------------------------ WMMA GEMM ---------------------------------
// C = act(A @ Bt^T + bias) + resid.  A: Mrows x K (f16 row-major),
// Bt: Ncols x K (f16 row-major, i.e. B stored K-contiguous).
// One wave computes a 32x64 strip. Optional f32 and/or f16 outputs.
// act: 0=none 1=relu 2=exp 3=leaky-relu(0.1)
__global__ void wmma_gemm(const _Float16* __restrict__ A, long sAb, int lda,
                          const _Float16* __restrict__ Bt, long sBb, int ldb,
                          const float* __restrict__ bias,
                          const float* __restrict__ resid, long sRb, int ldr,
                          float* __restrict__ C32, long sCb, int ldc,
                          _Float16* __restrict__ C16, long sC16b, int ldc16,
                          int Mrows, int Ncols, int K, int act)
{
    const int lane = threadIdx.x;        // 0..31 (wave32)
    const int m    = lane & 15;
    const int kh   = lane >> 4;          // lane half
    const int row0    = blockIdx.y << 5; // 2 row tiles of 16
    const int colBase = blockIdx.x << 6; // 4 column tiles of 16
    const long bz  = blockIdx.z;
    const _Float16* Ab = A  + bz * sAb;
    const _Float16* Bb = Bt + bz * sBb;

    const int  ar0 = row0 + m;           // A rows owned by this lane
    const int  ar1 = row0 + 16 + m;
    const bool hot = (row0 + 32 <= Mrows) && (colBase + 64 <= Ncols) &&
                     ((lda & 7) == 0) && ((ldb & 7) == 0);
    const int  K32 = K & ~31;

    v8f acc00 = {}, acc01 = {}, acc02 = {}, acc03 = {};   // row tile 0
    v8f acc10 = {}, acc11 = {}, acc12 = {}, acc13 = {};   // row tile 1

    // Guarded fallback for edge blocks and the K tail (cold path).
    auto generic_chunk = [&](int k0) {
        v16h af0, af1;
#pragma unroll
        for (int v = 0; v < 8; ++v) {
            int ka = (kh << 4) + (v << 1);
#pragma unroll
            for (int h2 = 0; h2 < 2; ++h2) {
                int k = k0 + ka + h2;
                af0[2 * v + h2] = (ar0 < Mrows && k < K) ? Ab[(long)ar0 * lda + k]
                                                         : (_Float16)0.0f;
                af1[2 * v + h2] = (ar1 < Mrows && k < K) ? Ab[(long)ar1 * lda + k]
                                                         : (_Float16)0.0f;
            }
        }
        v8f* a0p[4] = { &acc00, &acc01, &acc02, &acc03 };
        v8f* a1p[4] = { &acc10, &acc11, &acc12, &acc13 };
#pragma unroll
        for (int t = 0; t < 4; ++t) {
            const int bn = colBase + (t << 4) + m;
            v16h bf;
#pragma unroll
            for (int j = 0; j < 16; ++j) {
                int k = k0 + (kh << 4) + j;
                bf[j] = (k < K && bn < Ncols) ? Bb[(long)bn * ldb + k]
                                              : (_Float16)0.0f;
            }
            *a0p[t] = __builtin_amdgcn_wmma_f32_16x16x32_f16(false, af0, false, bf,
                                                             (short)0, *a0p[t], false, false);
            *a1p[t] = __builtin_amdgcn_wmma_f32_16x16x32_f16(false, af1, false, bf,
                                                             (short)0, *a1p[t], false, false);
        }
    };

    if (hot) {
        const _Float16* arow0 = Ab + (long)ar0 * lda + (kh << 4);
        const _Float16* arow1 = Ab + (long)ar1 * lda + (kh << 4);
        const _Float16* b0r = Bb + (long)(colBase +  0 + m) * ldb + (kh << 4);
        const _Float16* b1r = Bb + (long)(colBase + 16 + m) * ldb + (kh << 4);
        const _Float16* b2r = Bb + (long)(colBase + 32 + m) * ldb + (kh << 4);
        const _Float16* b3r = Bb + (long)(colBase + 48 + m) * ldb + (kh << 4);
        for (int k0 = 0; k0 < K32; k0 += 32) {
            // A fragments: two contiguous 8-half (16B) runs per lane per row.
            v8h a00 = *(const v8h*)(arow0 + k0);
            v8h a01 = *(const v8h*)(arow0 + k0 + 8);
            v8h a10 = *(const v8h*)(arow1 + k0);
            v8h a11 = *(const v8h*)(arow1 + k0 + 8);
            v16h af0 = __builtin_shufflevector(a00, a01, 0,1,2,3,4,5,6,7,
                                               8,9,10,11,12,13,14,15);
            v16h af1 = __builtin_shufflevector(a10, a11, 0,1,2,3,4,5,6,7,
                                               8,9,10,11,12,13,14,15);
            if (k0 + 32 < K32) {
                __builtin_prefetch(arow0 + k0 + 32, 0, 1);
                __builtin_prefetch(arow1 + k0 + 32, 0, 1);
            }
#pragma unroll
            for (int t = 0; t < 4; ++t) {
                const _Float16* br = (t == 0) ? b0r : (t == 1) ? b1r
                                   : (t == 2) ? b2r : b3r;
                v8h bb0 = *(const v8h*)(br + k0);
                v8h bb1 = *(const v8h*)(br + k0 + 8);
                v16h bf = __builtin_shufflevector(bb0, bb1, 0,1,2,3,4,5,6,7,
                                                  8,9,10,11,12,13,14,15);
                if (t == 0) { acc00 = __builtin_amdgcn_wmma_f32_16x16x32_f16(false, af0, false, bf, (short)0, acc00, false, false);
                              acc10 = __builtin_amdgcn_wmma_f32_16x16x32_f16(false, af1, false, bf, (short)0, acc10, false, false); }
                if (t == 1) { acc01 = __builtin_amdgcn_wmma_f32_16x16x32_f16(false, af0, false, bf, (short)0, acc01, false, false);
                              acc11 = __builtin_amdgcn_wmma_f32_16x16x32_f16(false, af1, false, bf, (short)0, acc11, false, false); }
                if (t == 2) { acc02 = __builtin_amdgcn_wmma_f32_16x16x32_f16(false, af0, false, bf, (short)0, acc02, false, false);
                              acc12 = __builtin_amdgcn_wmma_f32_16x16x32_f16(false, af1, false, bf, (short)0, acc12, false, false); }
                if (t == 3) { acc03 = __builtin_amdgcn_wmma_f32_16x16x32_f16(false, af0, false, bf, (short)0, acc03, false, false);
                              acc13 = __builtin_amdgcn_wmma_f32_16x16x32_f16(false, af1, false, bf, (short)0, acc13, false, false); }
            }
        }
        if (K32 < K) generic_chunk(K32);
    } else {
        for (int k0 = 0; k0 < K; k0 += 32) generic_chunk(k0);
    }

    // ---- epilogue: bias, activation, residual, store ----------------------
    auto store_tile = [&](const v8f& a, int rbase, int cb) {
        const int bn = cb + m;
        if (bn >= Ncols) return;
        float bv = bias ? bias[bn] : 0.0f;
#pragma unroll
        for (int v = 0; v < 8; ++v) {
            int r = rbase + v + (kh << 3);  // C/D layout: lanes 16-31 -> M+8
            if (r < Mrows) {
                float o = a[v] + bv;
                if (act == 1)      o = fmaxf(o, 0.0f);
                else if (act == 2) o = expf(o);
                else if (act == 3) o = (o > 0.0f) ? o : 0.1f * o;
                if (resid) o += resid[bz * sRb + (long)r * ldr + bn];
                if (C32) C32[bz * sCb + (long)r * ldc + bn] = o;
                if (C16) C16[bz * sC16b + (long)r * ldc16 + bn] = (_Float16)o;
            }
        }
    };
    store_tile(acc00, row0,      colBase);
    store_tile(acc01, row0,      colBase + 16);
    store_tile(acc02, row0,      colBase + 32);
    store_tile(acc03, row0,      colBase + 48);
    store_tile(acc10, row0 + 16, colBase);
    store_tile(acc11, row0 + 16, colBase + 16);
    store_tile(acc12, row0 + 16, colBase + 32);
    store_tile(acc13, row0 + 16, colBase + 48);
}

// --------------------- attention mask + row-normalize ----------------------
// scores holds exp(h.x^T) (f32). a = (e + 1e-5*I)*adj (adj diag=1), row-
// normalized; result written as f16 for the a@x GEMM.
__global__ void attn_norm_kernel(float* __restrict__ scores,
                                 const float* __restrict__ contacts,
                                 _Float16* __restrict__ ah)
{
    int i = blockIdx.x, b = blockIdx.y;
    float* row = scores + ((long)b * kN + i) * kN;
    const float* crow = contacts + ((long)b * kN + i) * kN;
    _Float16* hrow = ah + ((long)b * kN + i) * kN;
    __shared__ float red[256];
    float s = 0.0f;
    for (int k = threadIdx.x; k < kN; k += blockDim.x) {
        float e   = row[k];
        float adj = (k == i) ? 1.0f : crow[k];
        float a   = (e + ((k == i) ? 1e-5f : 0.0f)) * adj;
        row[k] = a;
        s += a;
    }
    red[threadIdx.x] = s;
    __syncthreads();
    for (int off = 128; off > 0; off >>= 1) {
        if (threadIdx.x < off) red[threadIdx.x] += red[threadIdx.x + off];
        __syncthreads();
    }
    float inv = 1.0f / red[0];
    for (int k = threadIdx.x; k < kN; k += blockDim.x)
        hrow[k] = (_Float16)(row[k] * inv);
}

// ------------------- CI = tanh(<u,v>_D) + 1 per (b,i) ----------------------
__global__ void dot_tanh_kernel(const float* __restrict__ U,
                                const float* __restrict__ V,
                                float* __restrict__ CI)
{
    long base = (long)blockIdx.x * kD;
    __shared__ float red[256];
    red[threadIdx.x] = U[base + threadIdx.x] * V[base + threadIdx.x];
    __syncthreads();
    for (int off = 128; off > 0; off >>= 1) {
        if (threadIdx.x < off) red[threadIdx.x] += red[threadIdx.x + off];
        __syncthreads();
    }
    if (threadIdx.x == 0) CI[blockIdx.x] = tanhf(red[0]) + 1.0f;
}

// ------- cat(f16) = [seq*CI0 , x_graph*CI1]  (B*N x 2D, GEMM A operand) ----
__global__ void concat_scale_kernel(const float* __restrict__ seq,
                                    const float* __restrict__ xg,
                                    const float* __restrict__ CI0,
                                    const float* __restrict__ CI1,
                                    _Float16* __restrict__ cat)
{
    int bi = blockIdx.x, d = threadIdx.x;
    float c0 = CI0[bi], c1 = CI1[bi];
    cat[(long)bi * 2 * kD + d]      = (_Float16)(seq[(long)bi * kD + d] * c0);
    cat[(long)bi * 2 * kD + kD + d] = (_Float16)(xg [(long)bi * kD + d] * c1);
}

// ----------- global sum of inter per batch (raw is ld=64 padded) -----------
__global__ void inter_sum_kernel(const float* __restrict__ raw,
                                 float* __restrict__ sums)
{
    int b = blockIdx.x;
    const float* p = raw + (long)b * kN * kMp;
    __shared__ float red[256];
    float s = 0.0f;
    for (int idx = threadIdx.x; idx < kN * kM; idx += blockDim.x) {
        int i = idx / kM, k = idx - i * kM;
        s += p[i * kMp + k];
    }
    red[threadIdx.x] = s;
    __syncthreads();
    for (int off = 128; off > 0; off >>= 1) {
        if (threadIdx.x < off) red[threadIdx.x] += red[threadIdx.x + off];
        __syncthreads();
    }
    if (threadIdx.x == 0) sums[b] = red[0];
}

__global__ void inter_norm_kernel(const float* __restrict__ raw,
                                  const float* __restrict__ sums,
                                  float* __restrict__ out)
{
    long g = (long)blockIdx.x * blockDim.x + threadIdx.x;
    if (g < (long)kB * kN * kM) {
        int  b   = (int)(g / (kN * kM));
        long rem = g - (long)b * kN * kM;
        int  i   = (int)(rem / kM);
        int  k   = (int)(rem - (long)i * kM);
        out[g] = raw[(long)b * kN * kMp + (long)i * kMp + k] / sums[b];
    }
}

// -------- cp[b,d] = sum_{i,k} tanh(pe[b,i,d]*ce[b,k,d]) * inter[b,i,k] -----
__global__ void cp_part_kernel(const float* __restrict__ pe,
                               const float* __restrict__ ce,
                               const float* __restrict__ inter,
                               float* __restrict__ part)
{
    int bi = blockIdx.x;
    int b  = bi / kN;
    int d  = threadIdx.x;
    float pv = pe[(long)bi * kD + d];
    const float* cb = ce + (long)b * kM * kD;
    const float* ib = inter + (long)bi * kM;
    float acc = 0.0f;
    for (int k = 0; k < kM; ++k)
        acc += tanhf(pv * cb[k * kD + d]) * ib[k];
    part[(long)bi * kD + d] = acc;
}

__global__ void cp_reduce_kernel(const float* __restrict__ part,
                                 float* __restrict__ cp)
{
    int b = blockIdx.x, d = threadIdx.x;
    float s = 0.0f;
    for (int i = 0; i < kN; ++i)
        s += part[((long)b * kN + i) * kD + d];
    cp[b * kD + d] = s;
}

// -------------- conv1d(k=4,s=2,pad=1) + lrelu + maxpool(4) -----------------
__global__ void conv_pool_kernel(const float* __restrict__ cp,
                                 const float* __restrict__ w,
                                 const float* __restrict__ bias,
                                 _Float16* __restrict__ pooled)
{
    int b = blockIdx.x;
    for (int idx = threadIdx.x; idx < 64 * 32; idx += blockDim.x) {
        int o = idx / 32, s2 = idx % 32;
        float mx = -1e30f;
        for (int j2 = 0; j2 < 4; ++j2) {
            int t = s2 * 4 + j2;
            float acc = bias[o];
            for (int j = 0; j < 4; ++j) {
                int pos = 2 * t - 1 + j;
                float v = (pos >= 0 && pos < kD) ? cp[b * kD + pos] : 0.0f;
                acc += v * w[o * 4 + j];
            }
            acc = (acc > 0.0f) ? acc : 0.1f * acc;
            mx = fmaxf(mx, acc);
        }
        pooled[b * 2048 + idx] = (_Float16)mx;
    }
}

// ----------------------- final 300 -> 1 projection -------------------------
__global__ void fc3_kernel(const float* __restrict__ x,
                           const float* __restrict__ w,
                           const float* __restrict__ bias,
                           float* __restrict__ out)
{
    int b = blockIdx.x;
    __shared__ float red[256];
    float s = 0.0f;
    for (int k = threadIdx.x; k < 300; k += blockDim.x)
        s += x[b * 300 + k] * w[k];
    red[threadIdx.x] = s;
    __syncthreads();
    for (int off = 128; off > 0; off >>= 1) {
        if (threadIdx.x < off) red[threadIdx.x] += red[threadIdx.x + off];
        __syncthreads();
    }
    if (threadIdx.x == 0) out[b] = red[0] + bias[0];
}

// ---------------------------------------------------------------------------
// host-side GEMM dispatch helper
// ---------------------------------------------------------------------------
static void launch_gemm(hipStream_t s,
                        const _Float16* A, long sAb, int lda,
                        const _Float16* Bt, long sBb, int ldb,
                        const float* bias,
                        const float* resid, long sRb, int ldr,
                        float* C32, long sCb, int ldc,
                        _Float16* C16, long sC16b, int ldc16,
                        int M, int N, int K, int act, int batch)
{
    dim3 grid((N + 63) / 64, (M + 31) / 32, batch);
    dim3 blk(32, 1, 1);
    wmma_gemm<<<grid, blk, 0, s>>>(A, sAb, lda, Bt, sBb, ldb, bias,
                                   resid, sRb, ldr, C32, sCb, ldc,
                                   C16, sC16b, ldc16, M, N, K, act);
}

extern "C" void kernel_launch(void* const* d_in, const int* in_sizes, int n_in,
                              void* d_out, int out_size, void* d_ws, size_t ws_size,
                              hipStream_t stream)
{
    (void)in_sizes; (void)n_in; (void)out_size; (void)ws_size;

    const int*   prot     = (const int*)  d_in[0];
    const float* drug_ver = (const float*)d_in[1];   // (8,56,43)
    const float* drug_adj = (const float*)d_in[2];   // (8,56,56)
    const float* contacts = (const float*)d_in[3];   // (8,1000,1000)
    const float* emb      = (const float*)d_in[4];   // (24,256)
    const float* w_attn   = (const float*)d_in[5];   // (7,256,256)
    const float* b_attn   = (const float*)d_in[6];
    const float* gw0      = (const float*)d_in[7];
    const float* gb0      = (const float*)d_in[8];
    const float* gw1      = (const float*)d_in[9];
    const float* gb1      = (const float*)d_in[10];
    const float* gwf      = (const float*)d_in[11];
    const float* gbf      = (const float*)d_in[12];
    const float* ci0_w1   = (const float*)d_in[13];
    const float* ci0_b1   = (const float*)d_in[14];
    const float* ci0_w2   = (const float*)d_in[15];
    const float* ci0_b2   = (const float*)d_in[16];
    const float* ci1_w1   = (const float*)d_in[17];
    const float* ci1_b1   = (const float*)d_in[18];
    const float* ci1_w2   = (const float*)d_in[19];
    const float* ci1_b2   = (const float*)d_in[20];
    const float* ci_wl    = (const float*)d_in[21];  // (512,256)
    const float* ci_bl    = (const float*)d_in[22];
    const float* gcn_w0   = (const float*)d_in[23];  // (43,256)
    const float* gcn_b0   = (const float*)d_in[24];
    const float* gcn_w1   = (const float*)d_in[25];
    const float* gcn_b1   = (const float*)d_in[26];
    const float* gcn_w2   = (const float*)d_in[27];
    const float* gcn_b2   = (const float*)d_in[28];
    const float* gcn_wf   = (const float*)d_in[29];
    const float* gcn_bf   = (const float*)d_in[30];
    const float* jp_w     = (const float*)d_in[31];
    const float* jp_b     = (const float*)d_in[32];
    const float* jc_w     = (const float*)d_in[33];
    const float* jc_b     = (const float*)d_in[34];
    const float* conv_w   = (const float*)d_in[35];  // (64,1,4)
    const float* conv_b   = (const float*)d_in[36];
    const float* r1_w     = (const float*)d_in[37];  // (2048,600)
    const float* r1_b     = (const float*)d_in[38];
    const float* r2_w     = (const float*)d_in[39];  // (600,300)
    const float* r2_b     = (const float*)d_in[40];
    const float* r3_w     = (const float*)d_in[41];  // (300,1)
    const float* r3_b     = (const float*)d_in[42];

    const long BN  = (long)kB * kN;          // 8000
    const long BND = BN * kD;                // 2,048,000
    const long BM  = (long)kB * kM;          // 448
    const long DD  = (long)kD * kD;          // 65536

    // ---------------- float workspace ----------------
    float* fp = (float*)d_ws;
    auto falloc = [&](long n) { float* p = fp; fp += n; return p; };
    float* SEQ = falloc(BND);                // x_seq f32
    float* XA  = falloc(BND);                // x ping (resid chain)
    float* XB  = falloc(BND);                // x pong / cp partials
    float* G   = falloc(BND);                // x_graph f32
    float* T   = falloc(BND);                // CI u outputs / padded IRAW
    float* PE  = falloc(BND);                // prot embedding f32
    float* SC  = falloc((long)kB * kN * kN); // raw exp scores f32
    float* CI0 = falloc(BN);
    float* CI1 = falloc(BN);
    float* CE  = falloc(BM * kD);            // comp embedding f32
    float* SUM = falloc(64);
    float* CP  = falloc(kB * kD);
    float* F2  = falloc(kB * 300 + 4);
    float* PART = XB;                        // cp partials reuse XB
    float* IRAW = T;                         // padded raw inter reuses T

    // ---------------- f16 workspace ----------------
    _Float16* hp = (_Float16*)fp;
    auto halloc = [&](long n) { _Float16* p = hp; hp += n; return p; };
    _Float16* XH   = halloc(BND);            // x f16 (GEMM A / scores Bt)
    _Float16* XTH  = halloc(BND);            // x^T f16 (a@x Bt); later p f16
    _Float16* SEQH = halloc(BND);            // x_seq f16
    _Float16* HH   = halloc(BND);            // h f16 / t1 f16 / mlp temps
    _Float16* TH   = halloc(BND);            // x_att f16; later x_graph f16
    _Float16* SCH  = halloc((long)kB * kN * kN); // normalized attention f16; later CAT
    // transposed f16 weights
    _Float16* WAT  = halloc(7 * DD);
    _Float16* W0T  = halloc(7 * DD);
    _Float16* W1T  = halloc(7 * DD);
    _Float16* WFT  = halloc(DD);
    _Float16* CI0W1T = halloc(DD);
    _Float16* CI0W2T = halloc(DD);
    _Float16* CI1W1T = halloc(DD);
    _Float16* CI1W2T = halloc(DD);
    _Float16* CIWLT  = halloc(2 * DD);       // 256 x 512
    _Float16* GW0T   = halloc((long)kD * 43);
    _Float16* GW1T   = halloc(DD);
    _Float16* GW2T   = halloc(DD);
    _Float16* GWFT   = halloc(DD);
    _Float16* JPWT   = halloc(DD);
    _Float16* JCWT   = halloc(DD);
    _Float16* R1WT   = halloc(600L * 2048);
    _Float16* R2WT   = halloc(300L * 600);
    // drug-side f16
    _Float16* VERH = halloc(BM * 43 + 8);
    _Float16* ADJH = halloc((long)kB * kM * kM);
    _Float16* C0H  = halloc(BM * kD);
    _Float16* C0T  = halloc(BM * kD);
    _Float16* C1H  = halloc(BM * kD);
    _Float16* CEH  = halloc(BM * kD);
    _Float16* QH   = halloc((long)kB * kMp * kD);
    _Float16* POH  = halloc(kB * 2048);
    _Float16* F1H  = halloc(kB * 600);
    _Float16* CATH = SCH;                    // B*N*2D fits in SCH region
    _Float16* GH   = TH;                     // x_graph f16 reuses TH
    _Float16* RPH  = HH;                     // relu(PE) f16 reuses HH
    _Float16* PH   = XTH;                    // p f16 reuses XTH

    float* out_inter = (float*)d_out;                    // 8*1000*56
    float* out_affn  = (float*)d_out + (long)kB * kN * kM;

    dim3 b256(256);

    // 0) one-time weight transpose-converts (f32 KxN -> f16 NxK)
    for (int l = 0; l < 7; ++l) {
        cvt_t_kernel<<<dim3(kD), b256, 0, stream>>>(w_attn + l * DD, WAT + l * DD, kD, kD);
        cvt_t_kernel<<<dim3(kD), b256, 0, stream>>>(gw0    + l * DD, W0T + l * DD, kD, kD);
        cvt_t_kernel<<<dim3(kD), b256, 0, stream>>>(gw1    + l * DD, W1T + l * DD, kD, kD);
    }
    cvt_t_kernel<<<dim3(kD), b256, 0, stream>>>(gwf,    WFT,    kD, kD);
    cvt_t_kernel<<<dim3(kD), b256, 0, stream>>>(ci0_w1, CI0W1T, kD, kD);
    cvt_t_kernel<<<dim3(kD), b256, 0, stream>>>(ci0_w2, CI0W2T, kD, kD);
    cvt_t_kernel<<<dim3(kD), b256, 0, stream>>>(ci1_w1, CI1W1T, kD, kD);
    cvt_t_kernel<<<dim3(kD), b256, 0, stream>>>(ci1_w2, CI1W2T, kD, kD);
    cvt_t_kernel<<<dim3(kD), b256, 0, stream>>>(ci_wl,  CIWLT,  2 * kD, kD);
    cvt_t_kernel<<<dim3(kD), b256, 0, stream>>>(gcn_w0, GW0T,   43, kD);
    cvt_t_kernel<<<dim3(kD), b256, 0, stream>>>(gcn_w1, GW1T,   kD, kD);
    cvt_t_kernel<<<dim3(kD), b256, 0, stream>>>(gcn_w2, GW2T,   kD, kD);
    cvt_t_kernel<<<dim3(kD), b256, 0, stream>>>(gcn_wf, GWFT,   kD, kD);
    cvt_t_kernel<<<dim3(kD), b256, 0, stream>>>(jp_w,   JPWT,   kD, kD);
    cvt_t_kernel<<<dim3(kD), b256, 0, stream>>>(jc_w,   JCWT,   kD, kD);
    cvt_t_kernel<<<dim3(600), b256, 0, stream>>>(r1_w,  R1WT,   2048, 600);
    cvt_t_kernel<<<dim3(300), b256, 0, stream>>>(r2_w,  R2WT,   600, 300);
    cvt_kernel<<<dim3((unsigned)((BM * 43 + 255) / 256)), b256, 0, stream>>>(drug_ver, VERH, BM * 43);
    cvt_kernel<<<dim3((unsigned)((kB * kM * kM + 255) / 256)), b256, 0, stream>>>(drug_adj, ADJH, (long)kB * kM * kM);

    // 1) embedding lookup
    embed_kernel<<<dim3((unsigned)BN), b256, 0, stream>>>(prot, emb, SEQ, XA, SEQH, XH);

    // 2) 7 GAT layers
    float* xc = XA; float* xn = XB;
    for (int l = 0; l < 7; ++l) {
        // x^T f16 for the a@x B operand
        transpose_h_kernel<<<dim3((unsigned)BN), b256, 0, stream>>>(XH, XTH, kN);
        // h = x @ w_attn + b  -> f16
        launch_gemm(stream, XH, 0, kD, WAT + l * DD, 0, kD, b_attn + l * kD,
                    nullptr, 0, 0, nullptr, 0, 0, HH, 0, kD,
                    (int)BN, kD, kD, 0, 1);
        // scores = exp(h @ x^T) -> f32  (Bt = x row-major, K-contiguous)
        launch_gemm(stream, HH, (long)kN * kD, kD, XH, (long)kN * kD, kD, nullptr,
                    nullptr, 0, 0, SC, (long)kN * kN, kN, nullptr, 0, 0,
                    kN, kN, kD, 2, kB);
        // mask + row-normalize -> f16 attention matrix
        attn_norm_kernel<<<dim3(kN, kB), b256, 0, stream>>>(SC, contacts, SCH);
        // x_att = a @ x -> f16   (Bt = x^T, K=1000: main loop + tail chunk)
        launch_gemm(stream, SCH, (long)kN * kN, kN, XTH, (long)kN * kD, kN, nullptr,
                    nullptr, 0, 0, nullptr, 0, 0, TH, (long)kN * kD, kD,
                    kN, kD, kN, 0, kB);
        // t1 = relu(x_att @ w0 + b0) -> f16
        launch_gemm(stream, TH, 0, kD, W0T + l * DD, 0, kD, gb0 + l * kD,
                    nullptr, 0, 0, nullptr, 0, 0, HH, 0, kD,
                    (int)BN, kD, kD, 1, 1);
        // x_new = relu(t1 @ w1 + b1) + x -> f32 (resid chain) and f16
        launch_gemm(stream, HH, 0, kD, W1T + l * DD, 0, kD, gb1 + l * kD,
                    xc, 0, kD, xn, 0, kD, XH, 0, kD,
                    (int)BN, kD, kD, 1, 1);
        float* tmp = xc; xc = xn; xn = tmp;
    }

    // 3) x_graph = x @ gat_wf + gat_bf  (f32 + f16)
    launch_gemm(stream, XH, 0, kD, WFT, 0, kD, gbf, nullptr, 0, 0,
                G, 0, kD, GH, 0, kD, (int)BN, kD, kD, 0, 1);

    // 4) CI0 = tanh(<mlp(x_graph), x_seq>) + 1
    launch_gemm(stream, GH, 0, kD, CI0W1T, 0, kD, ci0_b1, nullptr, 0, 0,
                nullptr, 0, 0, HH, 0, kD, (int)BN, kD, kD, 1, 1);
    launch_gemm(stream, HH, 0, kD, CI0W2T, 0, kD, ci0_b2, nullptr, 0, 0,
                T, 0, kD, nullptr, 0, 0, (int)BN, kD, kD, 0, 1);
    dot_tanh_kernel<<<dim3((unsigned)BN), b256, 0, stream>>>(T, SEQ, CI0);

    // 5) CI1 = tanh(<mlp(x_seq), x_graph>) + 1
    launch_gemm(stream, SEQH, 0, kD, CI1W1T, 0, kD, ci1_b1, nullptr, 0, 0,
                nullptr, 0, 0, HH, 0, kD, (int)BN, kD, kD, 1, 1);
    launch_gemm(stream, HH, 0, kD, CI1W2T, 0, kD, ci1_b2, nullptr, 0, 0,
                T, 0, kD, nullptr, 0, 0, (int)BN, kD, kD, 0, 1);
    dot_tanh_kernel<<<dim3((unsigned)BN), b256, 0, stream>>>(T, G, CI1);

    // 6) cat(f16) = [seq*CI0, x_graph*CI1]; prot_emb = cat @ ci_wl + ci_bl
    concat_scale_kernel<<<dim3((unsigned)BN), b256, 0, stream>>>(SEQ, G, CI0, CI1, CATH);
    launch_gemm(stream, CATH, 0, 2 * kD, CIWLT, 0, 2 * kD, ci_bl, nullptr, 0, 0,
                PE, 0, kD, nullptr, 0, 0, (int)BN, kD, 2 * kD, 0, 1);

    // 7) drug GCN (3 layers) + final projection
    launch_gemm(stream, VERH, 0, 43, GW0T, 0, 43, gcn_b0, nullptr, 0, 0,
                nullptr, 0, 0, C0H, 0, kD, (int)BM, kD, 43, 0, 1);
    const _Float16* sW[3] = { GW1T, GW2T, GWFT };
    const float*    sB[3] = { gcn_b1, gcn_b2, gcn_bf };
    for (int s2 = 0; s2 < 3; ++s2) {
        transpose_h_kernel<<<dim3((unsigned)BM), b256, 0, stream>>>(C0H, C0T, kM);
        launch_gemm(stream, ADJH, (long)kM * kM, kM, C0T, (long)kM * kD, kM, nullptr,
                    nullptr, 0, 0, nullptr, 0, 0, C1H, (long)kM * kD, kD,
                    kM, kD, kM, 1, kB);
        if (s2 < 2)
            launch_gemm(stream, C1H, 0, kD, sW[s2], 0, kD, sB[s2], nullptr, 0, 0,
                        nullptr, 0, 0, C0H, 0, kD, (int)BM, kD, kD, 0, 1);
        else
            launch_gemm(stream, C1H, 0, kD, sW[s2], 0, kD, sB[s2], nullptr, 0, 0,
                        CE, 0, kD, nullptr, 0, 0, (int)BM, kD, kD, 0, 1);
    }

    // 8) p = relu(prot_emb) @ jp_w + jp_b ; q = relu(comp_emb) @ jc_w + jc_b
    relu_h_kernel<<<dim3((unsigned)((BND + 255) / 256)), b256, 0, stream>>>(PE, RPH, BND);
    launch_gemm(stream, RPH, 0, kD, JPWT, 0, kD, jp_b, nullptr, 0, 0,
                nullptr, 0, 0, PH, 0, kD, (int)BN, kD, kD, 0, 1);
    relu_h_kernel<<<dim3((unsigned)((BM * kD + 255) / 256)), b256, 0, stream>>>(CE, CEH, BM * kD);
    launch_gemm(stream, CEH, (long)kM * kD, kD, JCWT, 0, kD, jc_b, nullptr, 0, 0,
                nullptr, 0, 0, QH, (long)kMp * kD, kD, kM, kD, kD, 0, kB);

    // 9) inter = exp(p @ q^T): N padded to 64 so all column tiles are full.
    launch_gemm(stream, PH, (long)kN * kD, kD, QH, (long)kMp * kD, kD, nullptr,
                nullptr, 0, 0, IRAW, (long)kN * kMp, kMp, nullptr, 0, 0,
                kN, kMp, kD, 2, kB);
    inter_sum_kernel<<<dim3(kB), b256, 0, stream>>>(IRAW, SUM);
    {
        long tot = (long)kB * kN * kM;
        inter_norm_kernel<<<dim3((unsigned)((tot + 255) / 256)), b256, 0, stream>>>(
            IRAW, SUM, out_inter);
    }

    // 10) cp[b,d] = sum_{i,k} tanh(pe*ce) * inter  (two-stage, deterministic)
    cp_part_kernel<<<dim3((unsigned)BN), b256, 0, stream>>>(PE, CE, out_inter, PART);
    cp_reduce_kernel<<<dim3(kB), dim3(kD), 0, stream>>>(PART, CP);

    // 11) conv1d + lrelu + maxpool -> (B, 2048) f16
    conv_pool_kernel<<<dim3(kB), b256, 0, stream>>>(CP, conv_w, conv_b, POH);

    // 12) FC head: 2048->600->300->1 (lrelu, lrelu, none)
    launch_gemm(stream, POH, 0, 2048, R1WT, 0, 2048, r1_b, nullptr, 0, 0,
                nullptr, 0, 0, F1H, 0, 600, kB, 600, 2048, 3, 1);
    launch_gemm(stream, F1H, 0, 600, R2WT, 0, 600, r2_b, nullptr, 0, 0,
                F2, 0, 300, nullptr, 0, 0, kB, 300, 600, 3, 1);
    fc3_kernel<<<dim3(kB), b256, 0, stream>>>(F2, r3_w, r3_b, out_affn);
}
